// SegmentDecoder_85899346303
// MI455X (gfx1250) — matrix operand order repaired and checked by reading, not compiled, not run on hardware
//
#include <hip/hip_runtime.h>
#include <hip/hip_bf16.h>

// ---------------------------------------------------------------------------
// SegmentDecoder on MI455X (gfx1250, wave32).
//  * wih/out_w GEMMs hoisted out of the recurrence -> v_wmma_f32_16x16x32_bf16,
//    2x2 register blocking (4 WMMA / wave / K-chunk, A/B fragment reuse).
//  * All GEMM operands pre-converted to bf16 once; tile staging is pure data
//    movement: GLOBAL_LOAD_ASYNC_TO_LDS_B128 + s_wait_asynccnt double buffer.
//  * Recurrent whh scans: persistent kernels, device-wide atomic barrier per
//    step, h staged in LDS; whh (4 x 12.6 MB) stays L2-resident (192 MB L2).
// ---------------------------------------------------------------------------

typedef __attribute__((ext_vector_type(4)))  int            v4i;
typedef __attribute__((ext_vector_type(8)))  float          v8f;
typedef __attribute__((ext_vector_type(16))) __bf16         v16bf;
typedef __attribute__((ext_vector_type(4)))  unsigned short us4;
typedef __attribute__((ext_vector_type(8)))  unsigned short us8;
typedef __attribute__((ext_vector_type(16))) unsigned short us16;

#define S_LEN 1024
#define T_LEN 512
#define EMB   512
#define HID   1024
#define G3H   3072

#define SCAN_WGS 64   // H / 16 output elems per WG

#if defined(__AMDGCN__) && \
    __has_builtin(__builtin_amdgcn_global_load_async_to_lds_b128) && \
    __has_builtin(__builtin_amdgcn_s_wait_asynccnt)
#define USE_ASYNC 1
#else
#define USE_ASYNC 0
#endif

__device__ __forceinline__ unsigned short f2bf(float f) {
  unsigned int u = __builtin_bit_cast(unsigned int, f);
  unsigned int r = u + 0x7FFFu + ((u >> 16) & 1u);   // round-to-nearest-even
  return (unsigned short)(r >> 16);
}

// ---------------------------------------------------------------------------
__global__ void init_kernel(unsigned int* __restrict__ cnts,
                            float* __restrict__ labels) {
  int t = threadIdx.x;
  if (t < 8) cnts[t] = 0u;
  for (int i = t; i < 240; i += blockDim.x) labels[i] = 0.f;
}

// fp32 -> bf16 bulk conversion (weights; n4 = n/4 float4 groups)
__global__ void cvt_bf16_kernel(const float* __restrict__ src,
                                unsigned short* __restrict__ dst, int n4) {
  int i = blockIdx.x * blockDim.x + threadIdx.x;
  if (i >= n4) return;
  float4 v = reinterpret_cast<const float4*>(src)[i];
  us4 o; o[0] = f2bf(v.x); o[1] = f2bf(v.y); o[2] = f2bf(v.z); o[3] = f2bf(v.w);
  reinterpret_cast<us4*>(dst)[i] = o;
}

// ---------------------------------------------------------------------------
// embedding gathers -> bf16 activations
// ---------------------------------------------------------------------------
__global__ void gather_enc_kernel(const int* __restrict__ ids,
                                  const float* __restrict__ emb,
                                  unsigned short* __restrict__ out,
                                  int rows, int E) {
  int idx = blockIdx.x * blockDim.x + threadIdx.x;
  int vec = E >> 2;
  if (idx >= rows * vec) return;
  int r = idx / vec, c = idx - r * vec;
  int id = ids[r];
  float4 v = reinterpret_cast<const float4*>(emb + (size_t)id * E)[c];
  us4 o; o[0] = f2bf(v.x); o[1] = f2bf(v.y); o[2] = f2bf(v.z); o[3] = f2bf(v.w);
  reinterpret_cast<us4*>(out + (size_t)r * E)[c] = o;
}

__global__ void gather_dec_kernel(const int* __restrict__ target,
                                  const int* __restrict__ sos,
                                  const float* __restrict__ emb,
                                  unsigned short* __restrict__ out,
                                  int rows, int E) {
  int idx = blockIdx.x * blockDim.x + threadIdx.x;
  int vec = E >> 2;
  if (idx >= rows * vec) return;
  int t = idx / vec, c = idx - t * vec;
  int id = (t == 0) ? sos[0] : target[t - 1];
  float4 v = reinterpret_cast<const float4*>(emb + (size_t)id * E)[c];
  us4 o; o[0] = f2bf(v.x); o[1] = f2bf(v.y); o[2] = f2bf(v.z); o[3] = f2bf(v.w);
  reinterpret_cast<us4*>(out + (size_t)t * E)[c] = o;
}

// ---------------------------------------------------------------------------
// C[M,N] = A[M,K] @ W[N,K]^T + bias[N]; A,W bf16 (row-major, K contiguous),
// C fp32. WG: 256 thr / 8 waves, tile 64M x 128N, wave tile 32x32 (2x2 WMMA),
// K chunked by 32, double-buffered LDS fed by async global->LDS copies.
// grid = (N/128, M/64). M%64==0, N%128==0, K%32==0 at all call sites.
// ---------------------------------------------------------------------------
__global__ __launch_bounds__(256)
void gemm_bf16_wmma_kernel(const unsigned short* __restrict__ A,
                           const unsigned short* __restrict__ W,
                           const float* __restrict__ bias,
                           float* __restrict__ C, int M, int N, int K) {
  __shared__ __align__(16) unsigned short Abuf[2][64][32];
  __shared__ __align__(16) unsigned short Bbuf[2][128][32];
  const int tid  = threadIdx.x;
  const int lane = tid & 31;
  const int wave = tid >> 5;
  const int m0 = blockIdx.y * 64;
  const int n0 = blockIdx.x * 128;
  const int wm = (wave >> 2) * 32;   // 0 / 32
  const int wn = (wave & 3) * 32;    // 0 / 32 / 64 / 96

  // cooperative staging slots (16B per async op)
  const int ar = tid >> 2, ac = (tid & 3) * 8;    // A: 64x32, 1 op/thread
  const int br = tid >> 1, bc = (tid & 1) * 16;   // B: 128x32, 2 ops/thread

  const int nchunks = K >> 5;

  auto issue = [&](int chunk, int buf) {
    const int k0 = chunk << 5;
    const unsigned short* as = A + (size_t)(m0 + ar) * K + k0 + ac;
    const unsigned short* bs = W + (size_t)(n0 + br) * K + k0 + bc;
#if USE_ASYNC
    // builtin params are generic int4* (per hipcc diagnostic); backend derives
    // the global/LDS address spaces. Order: (global src, LDS dst, off, cpol).
    __builtin_amdgcn_global_load_async_to_lds_b128(
        (v4i*)as, (v4i*)&Abuf[buf][ar][ac], 0, 0);
    __builtin_amdgcn_global_load_async_to_lds_b128(
        (v4i*)bs, (v4i*)&Bbuf[buf][br][bc], 0, 0);
    __builtin_amdgcn_global_load_async_to_lds_b128(
        (v4i*)(bs + 8), (v4i*)&Bbuf[buf][br][bc + 8], 0, 0);
#else
    *reinterpret_cast<us8*>(&Abuf[buf][ar][ac])     = *reinterpret_cast<const us8*>(as);
    *reinterpret_cast<us8*>(&Bbuf[buf][br][bc])     = *reinterpret_cast<const us8*>(bs);
    *reinterpret_cast<us8*>(&Bbuf[buf][br][bc + 8]) = *reinterpret_cast<const us8*>(bs + 8);
#endif
  };

  v8f acc[2][2];
#pragma unroll
  for (int i = 0; i < 2; ++i)
#pragma unroll
    for (int j = 0; j < 2; ++j)
      acc[i][j] = v8f{0.f, 0.f, 0.f, 0.f, 0.f, 0.f, 0.f, 0.f};

  issue(0, 0);
  for (int c = 0; c < nchunks; ++c) {
    const int buf = c & 1;
    if (c + 1 < nchunks) {
      issue(c + 1, buf ^ 1);
#if USE_ASYNC
      __builtin_amdgcn_s_wait_asynccnt(3);   // chunk c landed; c+1 in flight
#endif
    } else {
#if USE_ASYNC
      __builtin_amdgcn_s_wait_asynccnt(0);
#endif
    }
    __syncthreads();

    // build fragments per CDNA5 16-bit A(16x32)/B(32x16) lane layouts
    const int half = lane >> 4;
    const int l15  = lane & 15;
    v16bf afr[2], bfr[2];
#pragma unroll
    for (int i = 0; i < 2; ++i) {
      us8 alo = *reinterpret_cast<const us8*>(&Abuf[buf][wm + i * 16 + l15][half * 8]);
      us8 ahi = *reinterpret_cast<const us8*>(&Abuf[buf][wm + i * 16 + l15][half * 8 + 16]);
      us8 blo = *reinterpret_cast<const us8*>(&Bbuf[buf][wn + i * 16 + l15][half * 8]);
      us8 bhi = *reinterpret_cast<const us8*>(&Bbuf[buf][wn + i * 16 + l15][half * 8 + 16]);
      us16 av, bv;
#pragma unroll
      for (int j = 0; j < 8; ++j) {
        av[j] = alo[j]; av[j + 8] = ahi[j];
        bv[j] = blo[j]; bv[j + 8] = bhi[j];
      }
      afr[i] = __builtin_bit_cast(v16bf, av);
      bfr[i] = __builtin_bit_cast(v16bf, bv);
    }
#pragma unroll
    for (int i = 0; i < 2; ++i)
#pragma unroll
      for (int j = 0; j < 2; ++j)
        acc[i][j] = __builtin_amdgcn_wmma_f32_16x16x32_bf16(
            false, afr[i], false, bfr[j], (short)0, acc[i][j], false, false);
    __syncthreads();
  }

  // C/D layout: VGPR v -> M = base + (lane>>4)*8 + v, N = base + (lane&15)
  const int half = lane >> 4;
  const int l15  = lane & 15;
#pragma unroll
  for (int j = 0; j < 2; ++j) {
    const int n = n0 + wn + j * 16 + l15;
    const float bn = bias ? bias[n] : 0.f;
#pragma unroll
    for (int i = 0; i < 2; ++i) {
#pragma unroll
      for (int v = 0; v < 8; ++v) {
        int m = m0 + wm + i * 16 + half * 8 + v;
        C[(size_t)m * N + n] = acc[i][j][v] + bn;
      }
    }
  }
}

// ---------------------------------------------------------------------------
// GRU recurrent scan (one layer). Persistent: SCAN_WGS x 256 threads.
// Each WG owns 16 hidden indices -> 48 gh rows; wave computes 6 rows/step.
// Also emits a bf16 shadow of hs for the downstream WMMA GEMM.
// ---------------------------------------------------------------------------
__global__ __launch_bounds__(256)
void gru_scan_kernel(const float* __restrict__ gx,   // Tsteps x 3H (incl. bih)
                     const float* __restrict__ whh,  // 3H x H
                     const float* __restrict__ bhh,  // 3H
                     const float* __restrict__ h0,   // H
                     float* __restrict__ hs,         // Tsteps x H
                     unsigned short* __restrict__ hs_bf, // Tsteps x H bf16
                     int Tsteps, unsigned int* __restrict__ cnt) {
  __shared__ __align__(16) float lds_h[HID];
  __shared__ float ghbuf[48];
  const int tid    = threadIdx.x;
  const int lane   = tid & 31;
  const int wave   = tid >> 5;
  const int base_i = blockIdx.x * 16;

  for (int t = 0; t < Tsteps; ++t) {
    const float* h = (t == 0) ? h0 : (hs + (size_t)(t - 1) * HID);
    reinterpret_cast<float4*>(lds_h)[tid] =
        reinterpret_cast<const float4*>(h)[tid];
    __syncthreads();

#pragma unroll
    for (int rr = 0; rr < 6; ++rr) {
      int lr   = wave * 6 + rr;                 // 0..47
      int gate = lr >> 4;
      int ii   = lr & 15;
      int row  = gate * HID + base_i + ii;
      const float4* wrow = reinterpret_cast<const float4*>(whh + (size_t)row * HID);
      const float4* hv   = reinterpret_cast<const float4*>(lds_h);
      float acc = 0.f;
#pragma unroll
      for (int kk = 0; kk < 8; ++kk) {
        float4 w4 = wrow[lane + kk * 32];
        float4 h4 = hv[lane + kk * 32];
        acc = fmaf(w4.x, h4.x, acc);
        acc = fmaf(w4.y, h4.y, acc);
        acc = fmaf(w4.z, h4.z, acc);
        acc = fmaf(w4.w, h4.w, acc);
      }
#pragma unroll
      for (int off = 16; off > 0; off >>= 1) acc += __shfl_xor(acc, off, 32);
      if (lane == 0) ghbuf[lr] = acc;
    }
    __syncthreads();

    if (tid < 16) {
      int i = base_i + tid;
      float gr = ghbuf[tid]      + bhh[i];
      float gz = ghbuf[16 + tid] + bhh[HID + i];
      float gn = ghbuf[32 + tid] + bhh[2 * HID + i];
      const float* gxt = gx + (size_t)t * G3H;
      float xr = gxt[i], xz = gxt[HID + i], xn = gxt[2 * HID + i];
      float r = 1.f / (1.f + __expf(-(xr + gr)));
      float z = 1.f / (1.f + __expf(-(xz + gz)));
      float n = tanhf(xn + r * gn);
      float hn = (1.f - z) * n + z * lds_h[i];
      hs[(size_t)t * HID + i]    = hn;
      hs_bf[(size_t)t * HID + i] = f2bf(hn);
    }
    __threadfence();
    __syncthreads();
    if (tid == 0) {
      __hip_atomic_fetch_add(cnt, 1u, __ATOMIC_ACQ_REL, __HIP_MEMORY_SCOPE_AGENT);
      unsigned int want = (unsigned int)SCAN_WGS * (unsigned int)(t + 1);
      while (__hip_atomic_load(cnt, __ATOMIC_ACQUIRE, __HIP_MEMORY_SCOPE_AGENT) < want)
        __builtin_amdgcn_s_sleep(1);
    }
    __syncthreads();
  }
}

// ---------------------------------------------------------------------------
__global__ void states_kernel(const float* __restrict__ hs0,
                              const float* __restrict__ hs1,
                              float* __restrict__ out, int Tsteps, int H) {
  int idx = blockIdx.x * blockDim.x + threadIdx.x;
  int vec = H >> 2;
  if (idx >= Tsteps * vec) return;
  int t = idx / vec, c = idx - t * vec;
  float4* o = reinterpret_cast<float4*>(out + (size_t)t * 2 * H);
  o[c]       = reinterpret_cast<const float4*>(hs0 + (size_t)t * H)[c];
  o[vec + c] = reinterpret_cast<const float4*>(hs1 + (size_t)t * H)[c];
}

// ---------------------------------------------------------------------------
__global__ __launch_bounds__(256)
void cls_kernel(const float* __restrict__ last,
                const float* __restrict__ w0, const float* __restrict__ b0,
                const float* __restrict__ w1, const float* __restrict__ b1,
                const float* __restrict__ w2, const float* __restrict__ b2,
                float* __restrict__ out) {
  const int lane = threadIdx.x & 31;
  const int wave = threadIdx.x >> 5;
  for (int row = wave; row < 80; row += 8) {
    const float* w; const float* b; int obase;
    if (row < 50)      { int r = row;      w = w0 + (size_t)r * HID; b = b0 + r; obase = r; }
    else if (row < 70) { int r = row - 50; w = w1 + (size_t)r * HID; b = b1 + r; obase = 150 + r; }
    else               { int r = row - 70; w = w2 + (size_t)r * HID; b = b2 + r; obase = 210 + r; }
    float acc = 0.f;
    for (int k = lane; k < HID; k += 32) acc = fmaf(w[k], last[k], acc);
#pragma unroll
    for (int off = 16; off > 0; off >>= 1) acc += __shfl_xor(acc, off, 32);
    if (lane == 0) out[obase] = acc + b[0];
  }
}

// ---------------------------------------------------------------------------
extern "C" void kernel_launch(void* const* d_in, const int* in_sizes, int n_in,
                              void* d_out, int out_size, void* d_ws, size_t ws_size,
                              hipStream_t stream) {
  (void)in_sizes; (void)n_in; (void)out_size; (void)ws_size;
  const int*   char_seq = (const int*)d_in[0];
  const int*   tgt_seq  = (const int*)d_in[1];
  const float* enc_st   = (const float*)d_in[2];
  const float* emb      = (const float*)d_in[3];
  const float* e_wih0 = (const float*)d_in[4];  const float* e_whh0 = (const float*)d_in[5];
  const float* e_bih0 = (const float*)d_in[6];  const float* e_bhh0 = (const float*)d_in[7];
  const float* e_wih1 = (const float*)d_in[8];  const float* e_whh1 = (const float*)d_in[9];
  const float* e_bih1 = (const float*)d_in[10]; const float* e_bhh1 = (const float*)d_in[11];
  const float* d_wih0 = (const float*)d_in[12]; const float* d_whh0 = (const float*)d_in[13];
  const float* d_bih0 = (const float*)d_in[14]; const float* d_bhh0 = (const float*)d_in[15];
  const float* d_wih1 = (const float*)d_in[16]; const float* d_whh1 = (const float*)d_in[17];
  const float* d_bih1 = (const float*)d_in[18]; const float* d_bhh1 = (const float*)d_in[19];
  const float* out_w  = (const float*)d_in[20]; const float* out_b  = (const float*)d_in[21];
  const float* cw0 = (const float*)d_in[22]; const float* cb0 = (const float*)d_in[23];
  const float* cw1 = (const float*)d_in[24]; const float* cb1 = (const float*)d_in[25];
  const float* cw2 = (const float*)d_in[26]; const float* cb2 = (const float*)d_in[27];
  const int*   sos = (const int*)d_in[28];

  // --- workspace carve-out (256B-aligned blocks) ---
  char* wp = (char*)d_ws;
  auto carve = [&](size_t bytes) -> char* {
    char* r = wp; wp += (bytes + 255) & ~(size_t)255; return r;
  };
  float* gx_e0 = (float*)carve((size_t)S_LEN * G3H * 4);
  float* gx_e1 = (float*)carve((size_t)S_LEN * G3H * 4);
  float* gx_d0 = (float*)carve((size_t)T_LEN * G3H * 4);
  float* gx_d1 = (float*)carve((size_t)T_LEN * G3H * 4);
  float* hs_e0 = (float*)carve((size_t)S_LEN * HID * 4);
  float* hs_e1 = (float*)carve((size_t)S_LEN * HID * 4);
  float* hs_d0 = (float*)carve((size_t)T_LEN * HID * 4);
  float* hs_d1 = (float*)carve((size_t)T_LEN * HID * 4);
  unsigned short* encx_b  = (unsigned short*)carve((size_t)S_LEN * EMB * 2);
  unsigned short* decx_b  = (unsigned short*)carve((size_t)T_LEN * EMB * 2);
  unsigned short* hse0_b  = (unsigned short*)carve((size_t)S_LEN * HID * 2);
  unsigned short* hse1_b  = (unsigned short*)carve((size_t)S_LEN * HID * 2);
  unsigned short* hsd0_b  = (unsigned short*)carve((size_t)T_LEN * HID * 2);
  unsigned short* hsd1_b  = (unsigned short*)carve((size_t)T_LEN * HID * 2);
  unsigned short* ewih0_b = (unsigned short*)carve((size_t)G3H * EMB * 2);
  unsigned short* ewih1_b = (unsigned short*)carve((size_t)G3H * HID * 2);
  unsigned short* dwih0_b = (unsigned short*)carve((size_t)G3H * EMB * 2);
  unsigned short* dwih1_b = (unsigned short*)carve((size_t)G3H * HID * 2);
  unsigned short* outw_b  = (unsigned short*)carve((size_t)HID * HID * 2);
  unsigned int*   cnts    = (unsigned int*)carve(64);

  float* out_scores = (float*)d_out;                          // 512*1024
  float* out_states = out_scores + (size_t)T_LEN * HID;       // 512*2048
  float* out_labels = out_states + (size_t)T_LEN * 2 * HID;   // 240

  const int BT = 256;
  auto cvt = [&](const float* s, unsigned short* d, int n) {
    int n4 = n >> 2;
    hipLaunchKernelGGL(cvt_bf16_kernel, dim3((n4 + BT - 1) / BT), dim3(BT), 0, stream, s, d, n4);
  };

  // 0) reset barrier counters + zero padded label rows
  hipLaunchKernelGGL(init_kernel, dim3(1), dim3(256), 0, stream, cnts, out_labels);

  // 0b) one-shot bf16 conversion of all GEMM weights (amortized over M tiles)
  cvt(e_wih0, ewih0_b, G3H * EMB);
  cvt(e_wih1, ewih1_b, G3H * HID);
  cvt(d_wih0, dwih0_b, G3H * EMB);
  cvt(d_wih1, dwih1_b, G3H * HID);
  cvt(out_w,  outw_b,  HID * HID);

  // 1) encoder embedding gather (bf16)
  hipLaunchKernelGGL(gather_enc_kernel, dim3((S_LEN * (EMB / 4) + BT - 1) / BT),
                     dim3(BT), 0, stream, char_seq, emb, encx_b, S_LEN, EMB);
  // 2) gx enc L0 = enc_x @ wih0^T + bih0  (WMMA, async-LDS double buffered)
  hipLaunchKernelGGL(gemm_bf16_wmma_kernel, dim3(G3H / 128, S_LEN / 64), dim3(256), 0, stream,
                     encx_b, ewih0_b, e_bih0, gx_e0, S_LEN, G3H, EMB);
  // 3) scan enc L0
  hipLaunchKernelGGL(gru_scan_kernel, dim3(SCAN_WGS), dim3(256), 0, stream,
                     gx_e0, e_whh0, e_bhh0, enc_st, hs_e0, hse0_b, S_LEN, cnts + 0);
  // 4) gx enc L1
  hipLaunchKernelGGL(gemm_bf16_wmma_kernel, dim3(G3H / 128, S_LEN / 64), dim3(256), 0, stream,
                     hse0_b, ewih1_b, e_bih1, gx_e1, S_LEN, G3H, HID);
  // 5) scan enc L1
  hipLaunchKernelGGL(gru_scan_kernel, dim3(SCAN_WGS), dim3(256), 0, stream,
                     gx_e1, e_whh1, e_bhh1, enc_st + HID, hs_e1, hse1_b, S_LEN, cnts + 1);
  // 6) decoder input gather (sos shift, bf16)
  hipLaunchKernelGGL(gather_dec_kernel, dim3((T_LEN * (EMB / 4) + BT - 1) / BT),
                     dim3(BT), 0, stream, tgt_seq, sos, emb, decx_b, T_LEN, EMB);
  // 7) gx dec L0
  hipLaunchKernelGGL(gemm_bf16_wmma_kernel, dim3(G3H / 128, T_LEN / 64), dim3(256), 0, stream,
                     decx_b, dwih0_b, d_bih0, gx_d0, T_LEN, G3H, EMB);
  // 8) scan dec L0 (h0 = enc L0 final state)
  hipLaunchKernelGGL(gru_scan_kernel, dim3(SCAN_WGS), dim3(256), 0, stream,
                     gx_d0, d_whh0, d_bhh0, hs_e0 + (size_t)(S_LEN - 1) * HID,
                     hs_d0, hsd0_b, T_LEN, cnts + 2);
  // 9) gx dec L1
  hipLaunchKernelGGL(gemm_bf16_wmma_kernel, dim3(G3H / 128, T_LEN / 64), dim3(256), 0, stream,
                     hsd0_b, dwih1_b, d_bih1, gx_d1, T_LEN, G3H, HID);
  // 10) scan dec L1 (h0 = enc L1 final state)
  hipLaunchKernelGGL(gru_scan_kernel, dim3(SCAN_WGS), dim3(256), 0, stream,
                     gx_d1, d_whh1, d_bhh1, hs_e1 + (size_t)(S_LEN - 1) * HID,
                     hs_d1, hsd1_b, T_LEN, cnts + 3);
  // 11) char_scores = hs_d1 @ out_w^T + out_b -> d_out (WMMA)
  hipLaunchKernelGGL(gemm_bf16_wmma_kernel, dim3(HID / 128, T_LEN / 64), dim3(256), 0, stream,
                     hsd1_b, outw_b, out_b, out_scores, T_LEN, HID, HID);
  // 12) char_states interleave
  hipLaunchKernelGGL(states_kernel, dim3((T_LEN * (HID / 4) + BT - 1) / BT),
                     dim3(BT), 0, stream, hs_d0, hs_d1, out_states, T_LEN, HID);
  // 13) classifiers on char_scores[T-1]
  hipLaunchKernelGGL(cls_kernel, dim3(1), dim3(256), 0, stream,
                     out_scores + (size_t)(T_LEN - 1) * HID,
                     cw0, cb0, cw1, cb1, cw2, cb2, out_labels);
}
